// LongformerSelfAttentionForBart_70093866271378
// MI455X (gfx1250) — compile-verified, matrix operand order checked by reading, not run on hardware
//
#include <hip/hip_runtime.h>

#define SEQ 4096
#define BATCH 2
#define DM 1024
#define NH 16
#define HD 64
#define WH 256
#define NC 16

typedef float v8f __attribute__((ext_vector_type(8)));
typedef __bf16 v16bf __attribute__((ext_vector_type(16)));
typedef int v4i __attribute__((ext_vector_type(4)));

union Frag16 { uint4 u[2]; v16bf v; };

#if defined(__has_builtin)
#if __has_builtin(__builtin_amdgcn_global_load_async_to_lds_b128) && \
    __has_builtin(__builtin_amdgcn_s_wait_asynccnt)
#define HAVE_ASYNC 1
#endif
#endif
#ifndef HAVE_ASYNC
#define HAVE_ASYNC 0
#endif

// memory -> LDS, 16B per lane. Async (no VGPR round trip, ASYNCcnt) if the
// toolchain exposes the gfx1250 builtin; otherwise plain copy.
__device__ __forceinline__ void copy16_to_lds(const __bf16* g, __bf16* l) {
#if HAVE_ASYNC
  __builtin_amdgcn_global_load_async_to_lds_b128(
      (__attribute__((address_space(1))) v4i*)g,
      (__attribute__((address_space(3))) v4i*)l, 0, 0);
#else
  *reinterpret_cast<uint4*>(l) = *reinterpret_cast<const uint4*>(g);
#endif
}
__device__ __forceinline__ void async_copy_wait() {
#if HAVE_ASYNC
  __builtin_amdgcn_s_wait_asynccnt(0);
#endif
}

__device__ __forceinline__ v8f wmma_bf16(v16bf a, v16bf b, v8f c) {
  return __builtin_amdgcn_wmma_f32_16x16x32_bf16(false, a, false, b, (short)0, c,
                                                 false, false);
}

__device__ __forceinline__ float red_max16(float v) {
#pragma unroll
  for (int m = 1; m < 16; m <<= 1) v = fmaxf(v, __shfl_xor(v, m, 32));
  return v;
}
__device__ __forceinline__ float red_sum16(float v) {
#pragma unroll
  for (int m = 1; m < 16; m <<= 1) v += __shfl_xor(v, m, 32);
  return v;
}

// query (S,B,D) f32  ->  X (B*S, D) bf16
__global__ __launch_bounds__(256) void xconv(const float* __restrict__ q,
                                             __bf16* __restrict__ X) {
  int tid = blockIdx.x * 256 + threadIdx.x;
  int m = tid >> 8;
  int k0 = (tid & 255) << 2;
  int b = m >> 12, s = m & (SEQ - 1);
  float4 v = *reinterpret_cast<const float4*>(q + ((size_t)s * BATCH + b) * DM + k0);
  __bf16* dst = X + (size_t)m * DM + k0;
  dst[0] = (__bf16)v.x; dst[1] = (__bf16)v.y;
  dst[2] = (__bf16)v.z; dst[3] = (__bf16)v.w;
}

// W (K,N) f32 -> Wt (N,K) bf16  (tiled transpose through LDS)
__global__ __launch_bounds__(256) void wtrans(const float* __restrict__ Win,
                                              __bf16* __restrict__ Wt) {
  __shared__ float t[32][33];
  int x = threadIdx.x, y = threadIdx.y;  // 32 x 8
  int k0 = blockIdx.x * 32, n0 = blockIdx.y * 32;
#pragma unroll
  for (int j = 0; j < 4; ++j)
    t[y + j * 8][x] = Win[(size_t)(k0 + y + j * 8) * DM + n0 + x];
  __syncthreads();
#pragma unroll
  for (int j = 0; j < 4; ++j)
    Wt[(size_t)(n0 + y + j * 8) * DM + k0 + x] = (__bf16)t[x][y + j * 8];
}

// C = A(8192x1024) * Bt^T (+bias). MODE 0: q (scale log2e/8, head layout bf16)
// MODE 1: k/v (head layout bf16). MODE 2: final f32 to (S,B,D).
template <int MODE>
__global__ __launch_bounds__(256) void gemm_bf16(const __bf16* __restrict__ A,
                                                 const __bf16* __restrict__ Bt,
                                                 const float* __restrict__ bias,
                                                 __bf16* __restrict__ outb,
                                                 float* __restrict__ outf) {
  __shared__ __bf16 sB[64 * 64];  // [n][k], 8 KB, one 64-K stage
  const int bm = blockIdx.x * 128;
  const int bn = blockIdx.y * 64;
  const int wave = threadIdx.x >> 5;
  const int lane = threadIdx.x & 31;
  const int ln = lane & 15;
  const int kh = lane >> 4;
  const int mrow = bm + wave * 16 + ln;
  const __bf16* arow = A + (size_t)mrow * DM;

  v8f acc[4] = {};
  for (int k0 = 0; k0 < DM; k0 += 64) {
    __syncthreads();
    {  // stage B tile [64 n][64 k]: 32B per thread
      int n = threadIdx.x >> 2;
      int fg = (threadIdx.x & 3) * 16;
      const __bf16* src = Bt + (size_t)(bn + n) * DM + k0 + fg;
      __bf16* dst = &sB[n * 64 + fg];
      copy16_to_lds(src, dst);
      copy16_to_lds(src + 8, dst + 8);
    }
    async_copy_wait();
    __syncthreads();
#pragma unroll
    for (int ks = 0; ks < 64; ks += 32) {
      Frag16 a;
      a.u[0] = *reinterpret_cast<const uint4*>(arow + k0 + ks + kh * 8);
      a.u[1] = *reinterpret_cast<const uint4*>(arow + k0 + ks + 16 + kh * 8);
#pragma unroll
      for (int f = 0; f < 4; ++f) {
        Frag16 bf;
        const __bf16* bb = &sB[(f * 16 + ln) * 64 + ks];
        bf.u[0] = *reinterpret_cast<const uint4*>(bb + kh * 8);
        bf.u[1] = *reinterpret_cast<const uint4*>(bb + 16 + kh * 8);
        acc[f] = wmma_bf16(a.v, bf.v, acc[f]);
      }
    }
  }
#pragma unroll
  for (int f = 0; f < 4; ++f) {
    int n = bn + f * 16 + ln;
    float bv = bias[n];
#pragma unroll
    for (int r = 0; r < 8; ++r) {
      int m = bm + wave * 16 + r + 8 * kh;
      float val = acc[f][r] + bv;
      // q: 1/sqrt(HD) * log2(e) so attention works in the exp2 domain
      if (MODE == 0) val *= 0.18033688011112042f;
      int b = m >> 12, s = m & (SEQ - 1);
      if (MODE <= 1) {
        int h = n >> 6, d = n & (HD - 1);
        outb[((size_t)((b * NH + h) * SEQ + s)) * HD + d] = (__bf16)val;
      } else {
        outf[((size_t)(s * BATCH + b)) * DM + n] = val;
      }
    }
  }
}

// One block per (b,h,chunk). 16 waves, 1 q-tile (16 rows) each.
// Softmax bookkeeping batched over 64 keys per iteration; exp2 domain.
// Fully-masked slabs (block-uniform) and iterations (wave-uniform) skipped.
__global__ __launch_bounds__(512) void attn_kernel(const __bf16* __restrict__ qbuf,
                                                   const __bf16* __restrict__ kbuf,
                                                   const __bf16* __restrict__ vbuf,
                                                   __bf16* __restrict__ abuf) {
  __shared__ __bf16 sK[128 * 64];    // [key][feat]   16 KB
  __shared__ __bf16 sVt[64 * 128];   // [feat][key]   16 KB
  __shared__ __bf16 sP[16][16 * 64]; // per-wave P    32 KB

  const int bid = blockIdx.x;
  const int c = bid & (NC - 1);
  const int bh = bid >> 4;
  const int b = bh >> 4;
  const int h = bh & (NH - 1);
  const int tid = threadIdx.x;
  const int wave = tid >> 5;
  const int lane = tid & 31;
  const int ln = lane & 15;
  const int kh = lane >> 4;

  const size_t base = (size_t)bh * SEQ * HD;
  const __bf16* qb = qbuf + base;
  const __bf16* kb = kbuf + base;
  const __bf16* vb = vbuf + base;
  const int kwin0 = c * WH - WH;
  const int qrow0 = wave * 16;

  // live key-window range for this wave (band union over its 16 rows, and
  // the global kpos in [0,S) constraint). wave-uniform.
  const int lo = max(qrow0, -kwin0);
  const int hi = min(qrow0 + 15 + 2 * WH, SEQ - 1 - kwin0);

  Frag16 qf[2];
  {
    const __bf16* qr = qb + (size_t)(c * WH + qrow0 + ln) * HD;
#pragma unroll
    for (int fb = 0; fb < 2; ++fb) {
      qf[fb].u[0] = *reinterpret_cast<const uint4*>(qr + fb * 32 + kh * 8);
      qf[fb].u[1] = *reinterpret_cast<const uint4*>(qr + fb * 32 + 16 + kh * 8);
    }
  }

  const float ninf = -__builtin_inff();
  float mrow[8], lsum[8];
#pragma unroll
  for (int r = 0; r < 8; ++r) { mrow[r] = ninf; lsum[r] = 0.f; }
  v8f oacc[4] = {};

  for (int slab = 0; slab < 6; ++slab) {
    // block-uniform skip: slab entirely outside [0,S) in key positions
    const int sw0 = kwin0 + slab * 128;
    if (sw0 >= SEQ || sw0 + 127 < 0) continue;

    __syncthreads();
    {  // K slab: 128 keys x 64 feats, row-major (async to LDS)
      int kl = tid >> 2;
      int fg = (tid & 3) * 16;
      int kg = sw0 + kl;
      kg = kg < 0 ? 0 : (kg >= SEQ ? SEQ - 1 : kg);
      const __bf16* src = kb + (size_t)kg * HD + fg;
      __bf16* dst = &sK[kl * HD + fg];
      copy16_to_lds(src, dst);
      copy16_to_lds(src + 8, dst + 8);
    }
    {  // V slab transposed: [feat][key]
      int kp = tid & 63;
      int fg = (tid >> 6) * 8;
      int k0 = sw0 + kp * 2;
      int k1 = k0 + 1;
      k0 = k0 < 0 ? 0 : (k0 >= SEQ ? SEQ - 1 : k0);
      k1 = k1 < 0 ? 0 : (k1 >= SEQ ? SEQ - 1 : k1);
      uint4 v0 = *reinterpret_cast<const uint4*>(vb + (size_t)k0 * HD + fg);
      uint4 v1 = *reinterpret_cast<const uint4*>(vb + (size_t)k1 * HD + fg);
      const __bf16* e0 = reinterpret_cast<const __bf16*>(&v0);
      const __bf16* e1 = reinterpret_cast<const __bf16*>(&v1);
#pragma unroll
      for (int i = 0; i < 8; ++i) {
        sVt[(fg + i) * 128 + kp * 2] = e0[i];
        sVt[(fg + i) * 128 + kp * 2 + 1] = e1[i];
      }
    }
    async_copy_wait();
    __syncthreads();

#pragma unroll
    for (int it = 0; it < 2; ++it) {       // 64 keys per softmax iteration
      const int s0l = it * 64;
      const int kk0 = slab * 128 + s0l;
      // wave-uniform skip: iteration entirely outside this wave's live band
      if (kk0 > hi || kk0 + 63 < lo) continue;

      v8f sc[4] = {};
#pragma unroll
      for (int j = 0; j < 4; ++j)          // 4 key sub-tiles of 16
#pragma unroll
        for (int fb = 0; fb < 2; ++fb) {
          Frag16 bf;
          const __bf16* kr = &sK[(s0l + j * 16 + ln) * HD + fb * 32];
          bf.u[0] = *reinterpret_cast<const uint4*>(kr + kh * 8);
          bf.u[1] = *reinterpret_cast<const uint4*>(kr + 16 + kh * 8);
          sc[j] = wmma_bf16(qf[fb].v, bf.v, sc[j]);
        }
      int kk[4];
      bool kgok[4];
#pragma unroll
      for (int j = 0; j < 4; ++j) {
        kk[j] = kk0 + j * 16 + ln;
        kgok[j] = (unsigned)(kwin0 + kk[j]) < (unsigned)SEQ;
      }
      const int qbase = qrow0 + 8 * kh;
#pragma unroll
      for (int r = 0; r < 8; ++r) {
        const int qr = qbase + r;
        float sv[4];
#pragma unroll
        for (int j = 0; j < 4; ++j) {
          bool ok = ((unsigned)(kk[j] - qr) <= 2u * WH) && kgok[j];
          sv[j] = ok ? sc[j][r] : ninf;
        }
        float tmax = red_max16(fmaxf(fmaxf(sv[0], sv[1]), fmaxf(sv[2], sv[3])));
        float mnew = fmaxf(mrow[r], tmax);
        float alpha = (mrow[r] == mnew) ? 1.f : exp2f(mrow[r] - mnew);
        mrow[r] = mnew;
        float p[4];
#pragma unroll
        for (int j = 0; j < 4; ++j)
          p[j] = (sv[j] == ninf) ? 0.f : exp2f(sv[j] - mnew);
        lsum[r] = lsum[r] * alpha + red_sum16((p[0] + p[1]) + (p[2] + p[3]));
#pragma unroll
        for (int f = 0; f < 4; ++f) oacc[f][r] *= alpha;
#pragma unroll
        for (int j = 0; j < 4; ++j)
          sP[wave][(r + 8 * kh) * 64 + j * 16 + ln] = (__bf16)p[j];
      }
      asm volatile("" ::: "memory");  // keep per-wave LDS store->load order
      Frag16 pf[2];
      {
        const __bf16* pr = &sP[wave][ln * 64];
#pragma unroll
        for (int kf = 0; kf < 2; ++kf) {
          pf[kf].u[0] = *reinterpret_cast<const uint4*>(pr + kf * 32 + kh * 8);
          pf[kf].u[1] = *reinterpret_cast<const uint4*>(pr + kf * 32 + 16 + kh * 8);
        }
      }
#pragma unroll
      for (int f = 0; f < 4; ++f) {
        const __bf16* vr = &sVt[(f * 16 + ln) * 128 + s0l];
#pragma unroll
        for (int kf = 0; kf < 2; ++kf) {
          Frag16 vf;
          vf.u[0] = *reinterpret_cast<const uint4*>(vr + kf * 32 + kh * 8);
          vf.u[1] = *reinterpret_cast<const uint4*>(vr + kf * 32 + 16 + kh * 8);
          oacc[f] = wmma_bf16(pf[kf].v, vf.v, oacc[f]);
        }
      }
    }
  }

#pragma unroll
  for (int r = 0; r < 8; ++r) {
    float inv = 1.f / lsum[r];
    int sg = c * WH + qrow0 + r + 8 * kh;
    size_t row = (size_t)(b * SEQ + sg) * DM + (size_t)h * HD;
#pragma unroll
    for (int f = 0; f < 4; ++f)
      abuf[row + f * 16 + ln] = (__bf16)(oacc[f][r] * inv);
  }
}

extern "C" void kernel_launch(void* const* d_in, const int* in_sizes, int n_in,
                              void* d_out, int out_size, void* d_ws, size_t ws_size,
                              hipStream_t stream) {
  const float* query = (const float*)d_in[0];
  const float* Wq = (const float*)d_in[1];
  const float* bq = (const float*)d_in[2];
  const float* Wk = (const float*)d_in[3];
  const float* bk = (const float*)d_in[4];
  const float* Wv = (const float*)d_in[5];
  const float* bv = (const float*)d_in[6];
  const float* Wo = (const float*)d_in[7];
  const float* bo = (const float*)d_in[8];

  char* ws = (char*)d_ws;
  const size_t MB = 1024 * 1024;
  __bf16* Xbf = (__bf16*)(ws);             // 16 MiB  (B*S, D)
  __bf16* Wqt = (__bf16*)(ws + 16 * MB);   //  2 MiB each (N,K)
  __bf16* Wkt = (__bf16*)(ws + 18 * MB);
  __bf16* Wvt = (__bf16*)(ws + 20 * MB);
  __bf16* Wot = (__bf16*)(ws + 22 * MB);
  __bf16* qbuf = (__bf16*)(ws + 24 * MB);  // 16 MiB each, head-major
  __bf16* kbuf = (__bf16*)(ws + 40 * MB);
  __bf16* vbuf = (__bf16*)(ws + 56 * MB);
  __bf16* abuf = (__bf16*)(ws + 72 * MB);  // 16 MiB (B*S, D)

  xconv<<<8192, 256, 0, stream>>>(query, Xbf);
  wtrans<<<dim3(32, 32), dim3(32, 8), 0, stream>>>(Wq, Wqt);
  wtrans<<<dim3(32, 32), dim3(32, 8), 0, stream>>>(Wk, Wkt);
  wtrans<<<dim3(32, 32), dim3(32, 8), 0, stream>>>(Wv, Wvt);
  wtrans<<<dim3(32, 32), dim3(32, 8), 0, stream>>>(Wo, Wot);

  gemm_bf16<0><<<dim3(64, 16), 256, 0, stream>>>(Xbf, Wqt, bq, qbuf, nullptr);
  gemm_bf16<1><<<dim3(64, 16), 256, 0, stream>>>(Xbf, Wkt, bk, kbuf, nullptr);
  gemm_bf16<1><<<dim3(64, 16), 256, 0, stream>>>(Xbf, Wvt, bv, vbuf, nullptr);

  attn_kernel<<<BATCH * NH * NC, 512, 0, stream>>>(qbuf, kbuf, vbuf, abuf);

  gemm_bf16<2><<<dim3(64, 16), 256, 0, stream>>>(abuf, Wot, bo, nullptr,
                                                 (float*)d_out);
}
